// DuplexGAT_7670811590750
// MI455X (gfx1250) — matrix operand order repaired and checked by reading, not compiled, not run on hardware
//
#include <hip/hip_runtime.h>
#include <hip/hip_fp16.h>

typedef __attribute__((ext_vector_type(16))) _Float16 v16h;
typedef __attribute__((ext_vector_type(8)))  _Float16 v8h;
typedef __attribute__((ext_vector_type(8)))  float    v8f;
typedef int async_v4i __attribute__((vector_size(16)));

#define LM 30
#define LT 50
#define NN 80
#define FF 512
#define BB 1024

#if __has_builtin(__builtin_amdgcn_global_load_async_to_lds_b128) && \
    __has_builtin(__builtin_amdgcn_s_wait_asynccnt)
#define USE_ASYNC_LDS 1
#else
#define USE_ASYNC_LDS 0
#endif

__device__ __forceinline__ float gelu_exact(float v) {
    return 0.5f * v * (1.0f + erff(v * 0.7071067811865475f));
}

// Stage 16 bytes global -> LDS. Async path uses GLOBAL_LOAD_ASYNC_TO_LDS_B128
// (ASYNCcnt); fallback is an aligned v8h copy through VGPRs.
__device__ __forceinline__ void stage_copy16(const _Float16* g, _Float16* l) {
#if USE_ASYNC_LDS
    __builtin_amdgcn_global_load_async_to_lds_b128(
        (__attribute__((address_space(1))) async_v4i*)(void*)g,
        (__attribute__((address_space(3))) async_v4i*)(void*)l, 0, 0);
#else
    *(v8h*)l = *(const v8h*)g;
#endif
}

__device__ __forceinline__ void stage_wait() {
#if USE_ASYNC_LDS
    __builtin_amdgcn_s_wait_asynccnt(0);
#endif
}

// A fragment (16x32 f16) from an LDS row-major panel; row stride sA in halves.
// ISA 7.12.2 (16-bit A 16x32): lanes 0-15 hold M=0..15;
// VGPR i holds K = ((i&3)*2) + (i>=4?16:0) + 8*(lane>=16), two halves per VGPR.
// Per lane this is two contiguous 16-byte runs -> ds_load_b128 pairs.
__device__ __forceinline__ v16h frag_a_lds(const _Float16* As, int sA, int k0) {
    int lane = threadIdx.x & 31;
    int m = lane & 15, g = lane >> 4;
    v16h a;
#pragma unroll
    for (int i = 0; i < 8; ++i) {
        int kb = k0 + ((i & 4) ? 16 : 0) + g * 8 + (i & 3) * 2;
        a[2 * i]     = As[m * sA + kb];
        a[2 * i + 1] = As[m * sA + kb + 1];
    }
    return a;
}

// B fragment (32x16) from fragment-major packed f16 weights: for fragment
// (kt, nt), lane l's 16 halves are contiguous (32 bytes) -> global_load_b128 x2.
__device__ __forceinline__ v16h frag_b_pk(const v16h* __restrict__ Wp, int NT, int kt, int nt) {
    return Wp[((size_t)kt * NT + nt) * 32 + (threadIdx.x & 31)];
}

// ---------------- weight packing: f32 row-major [K][N] -> f16 fragment-major ----------------
__global__ void k_pack(const float* __restrict__ W, _Float16* __restrict__ P, int K, int N) {
    int p = blockIdx.x * blockDim.x + threadIdx.x;
    if (p >= K * N) return;
    int NT = N >> 4;
    int idx = p & 15, lane = (p >> 4) & 31, frag = p >> 9;
    int kt = frag / NT, nt = frag - kt * NT;
    int g = lane >> 4, i = idx >> 1, h = idx & 1;
    int k = kt * 32 + ((i & 4) ? 16 : 0) + 8 * g + 2 * (i & 3) + h;
    int n = nt * 16 + (lane & 15);
    P[p] = (_Float16)W[(size_t)k * N + n];
}

// eeT[l][t][f] = sum_h edge_type_emb[etype(t)][h] * gat_We[l][h][f]; t:0->etype0, 1->etype2
__global__ void k_eeT(const float* __restrict__ ete, const float* __restrict__ We,
                      float* __restrict__ eeT) {
    int i = blockIdx.x * blockDim.x + threadIdx.x;  // 2*2*512 = 2048
    if (i >= 2048) return;
    int f = i & 511, t = (i >> 9) & 1, l = i >> 10;
    int et = t ? 2 : 0;
    float s = 0.f;
#pragma unroll
    for (int h = 0; h < 4; ++h) s += ete[et * 4 + h] * We[(l * 4 + h) * 512 + f];
    eeT[i] = s;
}

// ---------------- input projection: x = gelu(ln(A@W + b)) ----------------
__global__ void __launch_bounds__(256) k1_inproj(
    const float* __restrict__ mirna, const float* __restrict__ target,
    const float* __restrict__ nte, const v16h* __restrict__ WP,
    const float* __restrict__ bias, const float* __restrict__ lng,
    const float* __restrict__ lnb, float* __restrict__ xout,
    _Float16* __restrict__ xhout) {
    __shared__ alignas(16) _Float16 As[16 * 264];
    __shared__ float Ys[16 * 520];
    int b = blockIdx.y;
    int m0 = blockIdx.x * 16;
    for (int i = threadIdx.x; i < 16 * 256; i += 256) {
        int r = i >> 8, k = i & 255;
        int node = m0 + r;
        float v;
        if (node < LM) v = mirna[((size_t)b * LM + node) * 256 + k] + nte[k];
        else           v = target[((size_t)b * LT + (node - LM)) * 256 + k] + nte[256 + k];
        As[r * 264 + k] = (_Float16)v;
    }
    __syncthreads();
    int wave = threadIdx.x >> 5, lane = threadIdx.x & 31;
    int n0 = wave * 64;
    const int NT = FF >> 4;  // 32
    v8f acc[4] = {};
#pragma unroll
    for (int kt = 0; kt < 8; ++kt) {
        v16h a = frag_a_lds(As, 264, kt * 32);
#pragma unroll
        for (int j = 0; j < 4; ++j) {
            v16h bf = frag_b_pk(WP, NT, kt, wave * 4 + j);
            acc[j] = __builtin_amdgcn_wmma_f32_16x16x32_f16(
                false, a, false, bf, (short)0, acc[j], false, false);
        }
    }
    int g = lane >> 4, nn = lane & 15;
#pragma unroll
    for (int j = 0; j < 4; ++j)
#pragma unroll
        for (int i = 0; i < 8; ++i) {
            int col = n0 + j * 16 + nn;
            Ys[(i + 8 * g) * 520 + col] = acc[j][i] + bias[col];
        }
    __syncthreads();
    int r = threadIdx.x >> 4, l16 = threadIdx.x & 15;
    float s = 0.f;
    for (int f = l16; f < FF; f += 16) s += Ys[r * 520 + f];
#pragma unroll
    for (int off = 8; off >= 1; off >>= 1) s += __shfl_xor(s, off, 32);
    float mu = s * (1.0f / FF);
    float vs = 0.f;
    for (int f = l16; f < FF; f += 16) { float d = Ys[r * 520 + f] - mu; vs += d * d; }
#pragma unroll
    for (int off = 8; off >= 1; off >>= 1) vs += __shfl_xor(vs, off, 32);
    float rstd = rsqrtf(vs * (1.0f / FF) + 1e-5f);
    size_t row = (size_t)b * NN + m0 + r;
    float* xo = xout + row * FF;
    _Float16* xho = xhout + row * FF;
    for (int f = l16; f < FF; f += 16) {
        float v = (Ys[r * 520 + f] - mu) * rstd * lng[f] + lnb[f];
        v = gelu_exact(v);
        xo[f] = v;
        xho[f] = (_Float16)v;
    }
}

// ---------------- per-layer dual GEMM, flat [81920 x 512] x [512 x 512] ----------------
// Block: 256 thr (8 waves, 4x2). Tile M=64, N=128. A (f16 activations) staged
// into double-buffered LDS via async global->LDS copies, pipelined vs WMMA.
__global__ void __launch_bounds__(256) k2_gemm(
    const _Float16* __restrict__ xh, const v16h* __restrict__ WlP,
    const v16h* __restrict__ WrP, const float* __restrict__ bl,
    const float* __restrict__ br, float* __restrict__ outl, float* __restrict__ outr) {
    __shared__ alignas(16) _Float16 As[2][64 * 136];
    int side = blockIdx.z;
    const v16h* Wp = side ? WrP : WlP;
    const float* bias = side ? br : bl;
    float* out = side ? outr : outl;
    size_t r0 = (size_t)blockIdx.x * 64;
    int n0blk = blockIdx.y * 128;
    int wave = threadIdx.x >> 5, lane = threadIdx.x & 31;
    int wm = wave & 3, wn = wave >> 2;
    int rowbase = wm * 16;
    int colbase = n0blk + wn * 64;
    const int NT = FF >> 4;  // 32
    v8f acc[4] = {};
    // issue chunk 0
    {
        const _Float16* src = xh + r0 * FF;
        for (int i = threadIdx.x * 8; i < 64 * 128; i += 2048) {
            int r = i >> 7, k = i & 127;
            stage_copy16(&src[(size_t)r * FF + k], &As[0][r * 136 + k]);
        }
    }
    for (int kc = 0; kc < 4; ++kc) {  // K chunks of 128
        int cur = kc & 1;
        stage_wait();
        __syncthreads();
        if (kc < 3) {  // prefetch next chunk into the other buffer
            const _Float16* src = xh + r0 * FF + (kc + 1) * 128;
            for (int i = threadIdx.x * 8; i < 64 * 128; i += 2048) {
                int r = i >> 7, k = i & 127;
                stage_copy16(&src[(size_t)r * FF + k], &As[cur ^ 1][r * 136 + k]);
            }
        }
#pragma unroll
        for (int ks = 0; ks < 4; ++ks) {
            v16h a = frag_a_lds(&As[cur][rowbase * 136], 136, ks * 32);
            int kt = kc * 4 + ks;
#pragma unroll
            for (int j = 0; j < 4; ++j) {
                v16h bf = frag_b_pk(Wp, NT, kt, (colbase >> 4) + j);
                acc[j] = __builtin_amdgcn_wmma_f32_16x16x32_f16(
                    false, a, false, bf, (short)0, acc[j], false, false);
            }
        }
    }
    int g = lane >> 4, nn = lane & 15;
#pragma unroll
    for (int j = 0; j < 4; ++j)
#pragma unroll
        for (int i = 0; i < 8; ++i) {
            int col = colbase + j * 16 + nn;
            out[(r0 + rowbase + i + 8 * g) * FF + col] = acc[j][i] + bias[col];
        }
}

// ---------------- attention + aggregate + residual + LN (+gelu) ----------------
// One workgroup per graph. Static topology: incoming(n) = {n±1, n±2 within chain} + self.
__global__ void __launch_bounds__(256) k3_attn(
    const float* __restrict__ xl, const float* __restrict__ xr,
    float* __restrict__ x, _Float16* __restrict__ xh,
    const float* __restrict__ eeT,      // [2][512] this layer
    const float* __restrict__ att,      // [4][128] this layer
    const float* __restrict__ bo, const float* __restrict__ g,
    const float* __restrict__ bln, int do_gelu) {
    extern __shared__ float sm[];
    float* xls  = sm;                 // 80*512
    float* satt = sm + NN * FF;       // 512
    float* see0 = satt + FF;          // 512
    float* see2 = see0 + FF;          // 512
    int b = blockIdx.x;
    const float* xlb = xl + (size_t)b * NN * FF;
    const float* xrb = xr + (size_t)b * NN * FF;
    float* xb = x + (size_t)b * NN * FF;
    _Float16* xhb = xh + (size_t)b * NN * FF;
    for (int i = threadIdx.x; i < NN * FF; i += 256) xls[i] = xlb[i];
    for (int i = threadIdx.x; i < FF; i += 256) {
        satt[i] = att[i]; see0[i] = eeT[i]; see2[i] = eeT[FF + i];
    }
    __syncthreads();
    int wave = threadIdx.x >> 5, lane = threadIdx.x & 31;
    for (int n = wave; n < NN; n += 8) {
        bool mir = n < LM;
        int base = mir ? 0 : LM;
        int len = mir ? LM : LT;
        int pos = n - base;
        int srcs[5]; float is2[5]; int ne = 0, c0 = 0, c2 = 0;
        const int offs[4] = {-1, 1, -2, 2};
#pragma unroll
        for (int t = 0; t < 4; ++t) {
            int p = pos + offs[t];
            if (p >= 0 && p < len) {
                srcs[ne] = base + p; is2[ne] = (t >= 2) ? 1.f : 0.f; ne++;
                if (t >= 2) c2++; else c0++;
            }
        }
        float inv = 1.0f / (float)(c0 + c2);
        float wc0 = (float)c0 * inv, wc2 = (float)c2 * inv;
        srcs[ne] = n; is2[ne] = -1.f;       // self loop: mean edge attr
        int nedges = ne + 1;
        float alpha[4][5];
        for (int h = 0; h < 4; ++h) {
            float xrv[4], attv[4], e0v[4], e2v[4], esv[4];
#pragma unroll
            for (int t = 0; t < 4; ++t) {
                int f = h * 128 + lane + 32 * t;
                xrv[t]  = xrb[(size_t)n * FF + f];
                attv[t] = satt[f];
                e0v[t]  = see0[f];
                e2v[t]  = see2[f];
                esv[t]  = wc0 * e0v[t] + wc2 * e2v[t];
            }
            float aa[5]; float amax = -3.4e38f;
            for (int e = 0; e < nedges; ++e) {
                const float* xs = &xls[srcs[e] * FF + h * 128];
                float s = 0.f;
#pragma unroll
                for (int t = 0; t < 4; ++t) {
                    float ee = (is2[e] < 0.f) ? esv[t] : (is2[e] > 0.f ? e2v[t] : e0v[t]);
                    float m = xs[lane + 32 * t] + xrv[t] + ee;
                    m = (m > 0.f) ? m : 0.2f * m;   // leaky_relu 0.2
                    s += m * attv[t];
                }
#pragma unroll
                for (int off = 16; off >= 1; off >>= 1) s += __shfl_xor(s, off, 32);
                aa[e] = s;
                amax = fmaxf(amax, s);
            }
            float den = 0.f;
            for (int e = 0; e < nedges; ++e) { aa[e] = expf(aa[e] - amax); den += aa[e]; }
            float rden = 1.0f / den;
            for (int e = 0; e < nedges; ++e) alpha[h][e] = aa[e] * rden;
        }
        float row[16];
        float mp = 0.f;
#pragma unroll
        for (int i = 0; i < 16; ++i) {
            int f = lane + 32 * i;
            int h = f >> 7;
            float o = 0.f;
            for (int e = 0; e < nedges; ++e) o += alpha[h][e] * xls[srcs[e] * FF + f];
            o += bo[f] + xb[(size_t)n * FF + f];
            row[i] = o; mp += o;
        }
#pragma unroll
        for (int off = 16; off >= 1; off >>= 1) mp += __shfl_xor(mp, off, 32);
        float mu = mp * (1.0f / FF);
        float vp = 0.f;
#pragma unroll
        for (int i = 0; i < 16; ++i) { float d = row[i] - mu; vp += d * d; }
#pragma unroll
        for (int off = 16; off >= 1; off >>= 1) vp += __shfl_xor(vp, off, 32);
        float rstd = rsqrtf(vp * (1.0f / FF) + 1e-5f);
#pragma unroll
        for (int i = 0; i < 16; ++i) {
            int f = lane + 32 * i;
            float v = (row[i] - mu) * rstd * g[f] + bln[f];
            if (do_gelu) v = gelu_exact(v);
            xb[(size_t)n * FF + f] = v;
            xhb[(size_t)n * FF + f] = (_Float16)v;
        }
    }
}

// ---------------- pooling: pooled[b] = [mean_n x, max_n x] ----------------
__global__ void __launch_bounds__(256) k4_pool(const float* __restrict__ x,
                                               float* __restrict__ pooled) {
    int b = blockIdx.x;
    for (int f = threadIdx.x; f < FF; f += 256) {
        float s = 0.f, mx = -3.4e38f;
        for (int n = 0; n < NN; ++n) {
            float v = x[((size_t)b * NN + n) * FF + f];
            s += v; mx = fmaxf(mx, v);
        }
        pooled[(size_t)b * (2 * FF) + f] = s * (1.0f / NN);
        pooled[(size_t)b * (2 * FF) + FF + f] = mx;
    }
}

// ---------------- output head: out = gelu(ln(pooled@W + b)) ----------------
__global__ void __launch_bounds__(256) k4_out(
    const float* __restrict__ pooled, const v16h* __restrict__ WP,
    const float* __restrict__ bias, const float* __restrict__ lng,
    const float* __restrict__ lnb, float* __restrict__ out) {
    __shared__ alignas(16) _Float16 As[16 * 1040];
    __shared__ float Ys[16 * 136];
    int m0 = blockIdx.x * 16;
    for (int i = threadIdx.x; i < 16 * 1024; i += 256) {
        int r = i >> 10, k = i & 1023;
        As[r * 1040 + k] = (_Float16)pooled[(size_t)(m0 + r) * 1024 + k];
    }
    __syncthreads();
    int wave = threadIdx.x >> 5, lane = threadIdx.x & 31;
    int n0 = wave * 16;
    v8f acc = {};
    for (int kt = 0; kt < 32; ++kt) {
        v16h a = frag_a_lds(As, 1040, kt * 32);
        v16h bf = frag_b_pk(WP, 8, kt, wave);
        acc = __builtin_amdgcn_wmma_f32_16x16x32_f16(
            false, a, false, bf, (short)0, acc, false, false);
    }
    int g = lane >> 4, nn = lane & 15;
#pragma unroll
    for (int i = 0; i < 8; ++i) {
        Ys[(i + 8 * g) * 136 + n0 + nn] = acc[i] + bias[n0 + nn];
    }
    __syncthreads();
    int r = threadIdx.x >> 4, l16 = threadIdx.x & 15;
    float s = 0.f;
    for (int f = l16; f < 128; f += 16) s += Ys[r * 136 + f];
#pragma unroll
    for (int off = 8; off >= 1; off >>= 1) s += __shfl_xor(s, off, 32);
    float mu = s * (1.0f / 128.0f);
    float vs = 0.f;
    for (int f = l16; f < 128; f += 16) { float d = Ys[r * 136 + f] - mu; vs += d * d; }
#pragma unroll
    for (int off = 8; off >= 1; off >>= 1) vs += __shfl_xor(vs, off, 32);
    float rstd = rsqrtf(vs * (1.0f / 128.0f) + 1e-5f);
    for (int f = l16; f < 128; f += 16) {
        float v = (Ys[r * 136 + f] - mu) * rstd * lng[f] + lnb[f];
        out[(size_t)(m0 + r) * 128 + f] = gelu_exact(v);
    }
}

// ---------------- host launcher ----------------
extern "C" void kernel_launch(void* const* d_in, const int* in_sizes, int n_in,
                              void* d_out, int out_size, void* d_ws, size_t ws_size,
                              hipStream_t stream) {
    (void)in_sizes; (void)n_in; (void)out_size; (void)ws_size;
    const float* mirna    = (const float*)d_in[0];
    const float* target   = (const float*)d_in[1];
    const float* nte      = (const float*)d_in[2];
    const float* ete      = (const float*)d_in[3];
    const float* in_W     = (const float*)d_in[4];
    const float* in_b     = (const float*)d_in[5];
    const float* in_ln_g  = (const float*)d_in[6];
    const float* in_ln_b  = (const float*)d_in[7];
    const float* gat_Wl   = (const float*)d_in[8];
    const float* gat_bl   = (const float*)d_in[9];
    const float* gat_Wr   = (const float*)d_in[10];
    const float* gat_br   = (const float*)d_in[11];
    const float* gat_We   = (const float*)d_in[12];
    const float* gat_att  = (const float*)d_in[13];
    const float* gat_bo   = (const float*)d_in[14];
    const float* ln_g     = (const float*)d_in[15];
    const float* ln_b     = (const float*)d_in[16];
    const float* out_W    = (const float*)d_in[17];
    const float* out_b    = (const float*)d_in[18];
    const float* out_ln_g = (const float*)d_in[19];
    const float* out_ln_b = (const float*)d_in[20];
    float* out = (float*)d_out;

    char* ws = (char*)d_ws;
    const size_t XBYTES = (size_t)BB * NN * FF * sizeof(float);  // 167,772,160
    _Float16* w16_inW  = (_Float16*)(ws + 0);
    _Float16* w16_Wl   = (_Float16*)(ws + 262144);
    _Float16* w16_Wr   = (_Float16*)(ws + 1310720);
    _Float16* w16_outW = (_Float16*)(ws + 2359296);
    float* eeT    = (float*)(ws + 2621440);
    float* x      = (float*)(ws + 2629632);
    float* xl     = (float*)(ws + 2629632 + XBYTES);
    float* xr     = (float*)(ws + 2629632 + 2 * XBYTES);
    float* pooled = (float*)(ws + 2629632 + 3 * XBYTES);
    _Float16* xh  = (_Float16*)(ws + 2629632 + 3 * XBYTES + 4194304);

    // pack weights into fragment-major f16
    k_pack<<<512, 256, 0, stream>>>(in_W, w16_inW, 256, 512);
    for (int l = 0; l < 2; ++l) {
        k_pack<<<1024, 256, 0, stream>>>(gat_Wl + (size_t)l * 512 * 512,
                                         w16_Wl + (size_t)l * 512 * 512, 512, 512);
        k_pack<<<1024, 256, 0, stream>>>(gat_Wr + (size_t)l * 512 * 512,
                                         w16_Wr + (size_t)l * 512 * 512, 512, 512);
    }
    k_pack<<<512, 256, 0, stream>>>(out_W, w16_outW, 1024, 128);
    k_eeT<<<8, 256, 0, stream>>>(ete, gat_We, eeT);

    k1_inproj<<<dim3(5, BB), 256, 0, stream>>>(mirna, target, nte,
                                               (const v16h*)w16_inW,
                                               in_b, in_ln_g, in_ln_b, x, xh);

    const size_t k3_smem = (size_t)(NN * FF + 3 * FF) * sizeof(float);  // ~170 KB (<320 KB/WGP)
    (void)hipFuncSetAttribute(reinterpret_cast<const void*>(k3_attn),
                              hipFuncAttributeMaxDynamicSharedMemorySize, (int)k3_smem);

    for (int l = 0; l < 2; ++l) {
        k2_gemm<<<dim3(1280, 4, 2), 256, 0, stream>>>(
            xh, (const v16h*)(w16_Wl + (size_t)l * 512 * 512),
            (const v16h*)(w16_Wr + (size_t)l * 512 * 512),
            gat_bl + l * 512, gat_br + l * 512, xl, xr);
        k3_attn<<<BB, 256, k3_smem, stream>>>(
            xl, xr, x, xh, eeT + l * 1024, gat_att + l * 512, gat_bo + l * 512,
            ln_g + l * 512, ln_b + l * 512, (l == 0) ? 1 : 0);
    }

    k4_pool<<<BB, 256, 0, stream>>>(x, pooled);
    k4_out<<<64, 256, 0, stream>>>(pooled, (const v16h*)w16_outW,
                                   out_b, out_ln_g, out_ln_b, out);
}
